// AttentionRNN_51634096833032
// MI455X (gfx1250) — compile-verified, matrix operand order checked by reading, not run on hardware
//
#include <hip/hip_runtime.h>
#include <cmath>

// ---------------------------------------------------------------------------
// AttentionRNN (Bahdanau attention) for gfx1250 / MI455X
//   phase 0a: swizzle w2 fp32 -> bf16 in WMMA B-fragment order       (2 MB, L2-resident)
//   phase 0b: qpb[b,u] = query@w1 + b1 + b2                          (tiny)
//   phase 1 : scores[b,t] = sum_u tanh(qpb + values@w2) * v[u]       (fused bf16 WMMA GEMM)
//             each wave owns all 4 M-subtiles (4 indep. accumulators) and a
//             unique N slot -> every B fragment loaded once per WG (2 MB min)
//   phase 2a: softmax over T, write attention weights, zero context
//   phase 2b: context[b,d] = sum_t attn * values (fp32 streaming, atomics)
// ---------------------------------------------------------------------------

typedef __attribute__((ext_vector_type(16))) __bf16 v16bf;
typedef __attribute__((ext_vector_type(8)))  float  v8f;

#define B_DIM 32
#define T_DIM 2048
#define D_DIM 1024
#define U_DIM 1024
#define TM    64          // t-rows per workgroup in phase 1
#define ROWU  516         // u32 per LDS row (512 data + 4 pad -> 4-bank stagger)

__device__ __forceinline__ unsigned int pack_bf16(float a, float b) {
  unsigned int ua = __float_as_uint(a);
  unsigned int ub = __float_as_uint(b);
  ua = (ua + 0x7FFFu + ((ua >> 16) & 1u)) >> 16;   // round-to-nearest-even
  ub = (ub + 0x7FFFu + ((ub >> 16) & 1u)) >> 16;
  return ua | (ub << 16);
}

// Branch-free tanh with correct saturation: tanh(x) = 1 - 2/(exp(2x)+1).
// e=inf -> rcp=0 -> 1 ; e=0 -> -1. Uses v_exp_f32 + v_rcp_f32.
__device__ __forceinline__ float fast_tanh(float x) {
  float e = __expf(2.0f * x);
  float r = __builtin_amdgcn_rcpf(e + 1.0f);
  return fmaf(-2.0f, r, 1.0f);
}

union ABFrag { uint4 u4[2]; v16bf v; };

// ---------------------------------------------------------------------------
// Phase 0a: w2 [D][U] fp32 -> swizzled bf16 fragments.
// Per (ktile, n) slot: 64 bytes = [K+0..7][K+16..23][K+8..15][K+24..31]
// so a lane's 32-byte B-fragment load is contiguous (lane<16 -> first 32B,
// lane>=16 -> second 32B), matching the 16-bit WMMA B VGPR layout.
// ---------------------------------------------------------------------------
__global__ __launch_bounds__(256)
void prep_w2_kernel(const float* __restrict__ w2, uint4* __restrict__ w2p) {
  int idx = blockIdx.x * 256 + threadIdx.x;       // idx = kt*1024 + n
  int kt  = idx >> 10;
  int n   = idx & 1023;
  const float* col = w2 + (size_t)kt * 32 * U_DIM + n;
  const int korder[4] = {0, 16, 8, 24};
  uint4* dst = w2p + (size_t)idx * 4;
#pragma unroll
  for (int q = 0; q < 4; ++q) {
    unsigned int p[4];
#pragma unroll
    for (int i = 0; i < 4; ++i) {
      int k = korder[q] + 2 * i;
      p[i] = pack_bf16(col[(size_t)k * U_DIM], col[(size_t)(k + 1) * U_DIM]);
    }
    dst[q] = make_uint4(p[0], p[1], p[2], p[3]);
  }
}

// ---------------------------------------------------------------------------
// Phase 0b: qpb[b][u] = query[b,:] @ w1[:,u] + b1[u] + b2[u]   (fp32)
// ---------------------------------------------------------------------------
__global__ __launch_bounds__(256)
void prep_qpb_kernel(const float* __restrict__ query, const float* __restrict__ w1,
                     const float* __restrict__ b1, const float* __restrict__ b2,
                     float* __restrict__ qpb) {
  int u = blockIdx.x * 256 + threadIdx.x;
  int b = blockIdx.y;
  const float* q = query + b * D_DIM;
  float acc = 0.0f;
  for (int d = 0; d < D_DIM; ++d)
    acc = fmaf(q[d], w1[(size_t)d * U_DIM + u], acc);
  qpb[b * U_DIM + u] = acc + b1[u] + b2[u];
}

// ---------------------------------------------------------------------------
// Phase 1: fused  scores[b,t] = sum_u tanh(qpb[b,u] + (values@w2)[b,t,u]) * v[u]
// One workgroup = 64 t-rows of one batch, 8 waves.
// wave = N slot (n0 = wave*16, stride 128). Each wave computes all 4 M
// subtiles per N tile: one B fragment load feeds 4 independent WMMA chains.
// ---------------------------------------------------------------------------
__global__ __launch_bounds__(256)
void attn_score_kernel(const float* __restrict__ values,
                       const uint4* __restrict__ w2p,
                       const float* __restrict__ qpb,
                       const float* __restrict__ vvec,
                       float* __restrict__ scores) {
  extern __shared__ char smem_raw[];
  unsigned int* lds_vals  = (unsigned int*)smem_raw;                    // TM * ROWU u32
  float*        lds_score = (float*)(smem_raw + (size_t)TM * ROWU * 4); // TM floats

  const int b   = blockIdx.y;
  const int t0  = blockIdx.x * TM;
  const int tid = threadIdx.x;

  if (tid < TM) lds_score[tid] = 0.0f;

  // Stage values[b, t0:t0+64, :] -> LDS as bf16 (each element converted once chip-wide)
  {
    const float2* src = (const float2*)(values + ((size_t)b * T_DIM + t0) * D_DIM);
    for (int i = tid; i < TM * 512; i += 256) {
      int r = i >> 9;
      int p = i & 511;
      float2 f = src[(size_t)r * 512 + p];
      lds_vals[r * ROWU + p] = pack_bf16(f.x, f.y);
    }
  }
  __syncthreads();

  const int wave  = tid >> 5;       // N slot 0..7
  const int lane  = tid & 31;
  const int lhalf = lane >> 4;      // 0: K{0..7,16..23}, 1: K{8..15,24..31}
  const int l16   = lane & 15;

  const unsigned int* arow0 = lds_vals + l16 * ROWU;   // M subtile j at +j*16 rows

  float svec[4][8] = {{0.f}};

  for (int n_it = 0; n_it < U_DIM / 128; ++n_it) {
    const int n = wave * 16 + n_it * 128 + l16;        // this lane's column
    v8f c0 = {}, c1 = {}, c2 = {}, c3 = {};
    const uint4* bp = w2p + ((size_t)n << 2) + (lhalf << 1);
#pragma unroll 2
    for (int kt = 0; kt < D_DIM / 32; ++kt) {
      ABFrag bf;
      bf.u4[0] = bp[0];
      bf.u4[1] = bp[1];
      bp += (size_t)U_DIM * 4;                         // next K tile
      const int ku = kt * 16 + lhalf * 4;
      ABFrag a0, a1, a2, a3;
      a0.u4[0] = *(const uint4*)(arow0 + ku);
      a0.u4[1] = *(const uint4*)(arow0 + ku + 8);
      a1.u4[0] = *(const uint4*)(arow0 + 16 * ROWU + ku);
      a1.u4[1] = *(const uint4*)(arow0 + 16 * ROWU + ku + 8);
      a2.u4[0] = *(const uint4*)(arow0 + 32 * ROWU + ku);
      a2.u4[1] = *(const uint4*)(arow0 + 32 * ROWU + ku + 8);
      a3.u4[0] = *(const uint4*)(arow0 + 48 * ROWU + ku);
      a3.u4[1] = *(const uint4*)(arow0 + 48 * ROWU + ku + 8);
      c0 = __builtin_amdgcn_wmma_f32_16x16x32_bf16(false, a0.v, false, bf.v, (short)0, c0, false, false);
      c1 = __builtin_amdgcn_wmma_f32_16x16x32_bf16(false, a1.v, false, bf.v, (short)0, c1, false, false);
      c2 = __builtin_amdgcn_wmma_f32_16x16x32_bf16(false, a2.v, false, bf.v, (short)0, c2, false, false);
      c3 = __builtin_amdgcn_wmma_f32_16x16x32_bf16(false, a3.v, false, bf.v, (short)0, c3, false, false);
    }
    // Epilogue: tanh + dot with v, accumulated per lane (lane owns column n)
    const float qn = qpb[b * U_DIM + n];
    const float vn = vvec[n];
#pragma unroll
    for (int r = 0; r < 8; ++r) {
      svec[0][r] += fast_tanh(qn + c0[r]) * vn;
      svec[1][r] += fast_tanh(qn + c1[r]) * vn;
      svec[2][r] += fast_tanh(qn + c2[r]) * vn;
      svec[3][r] += fast_tanh(qn + c3[r]) * vn;
    }
  }

  // Reduce over the 16 lanes (N columns) within each half-wave
#pragma unroll
  for (int off = 8; off >= 1; off >>= 1) {
#pragma unroll
    for (int j = 0; j < 4; ++j)
#pragma unroll
      for (int r = 0; r < 8; ++r)
        svec[j][r] += __shfl_xor(svec[j][r], off, 16);
  }
  if (l16 == 0) {
#pragma unroll
    for (int j = 0; j < 4; ++j)
#pragma unroll
      for (int r = 0; r < 8; ++r)
        atomicAdd(&lds_score[j * 16 + lhalf * 8 + r], svec[j][r]); // merge 8 N-slot waves
  }
  __syncthreads();

  if (tid < TM)
    scores[(size_t)b * T_DIM + t0 + tid] = lds_score[tid];
}

// ---------------------------------------------------------------------------
// Phase 2a: softmax over T per batch; writes attention weights to d_out and
// zeroes the context region (required before atomic accumulation).
// ---------------------------------------------------------------------------
__global__ __launch_bounds__(256)
void softmax_kernel(const float* __restrict__ scores, const float* __restrict__ bv,
                    float* __restrict__ out_ctx, float* __restrict__ out_attn) {
  const int b = blockIdx.x;
  const int tid = threadIdx.x;
  __shared__ float red[256];

  float loc[8];
  float mx = -INFINITY;
#pragma unroll
  for (int i = 0; i < 8; ++i) {
    loc[i] = scores[(size_t)b * T_DIM + tid + i * 256] + bv[0];
    mx = fmaxf(mx, loc[i]);
  }
  red[tid] = mx; __syncthreads();
  for (int s = 128; s >= 1; s >>= 1) {
    if (tid < s) red[tid] = fmaxf(red[tid], red[tid + s]);
    __syncthreads();
  }
  mx = red[0]; __syncthreads();

  float sum = 0.0f;
#pragma unroll
  for (int i = 0; i < 8; ++i) { loc[i] = __expf(loc[i] - mx); sum += loc[i]; }
  red[tid] = sum; __syncthreads();
  for (int s = 128; s >= 1; s >>= 1) {
    if (tid < s) red[tid] += red[tid + s];
    __syncthreads();
  }
  const float inv = 1.0f / red[0];
#pragma unroll
  for (int i = 0; i < 8; ++i)
    out_attn[(size_t)b * T_DIM + tid + i * 256] = loc[i] * inv;

  for (int i = tid; i < D_DIM; i += 256)
    out_ctx[(size_t)b * D_DIM + i] = 0.0f;
}

// ---------------------------------------------------------------------------
// Phase 2b: context[b,d] = sum_t attn[b,t] * values[b,t,d]  (fp32 streaming)
// grid (T/256, B); each block handles 256 t's, full D via float4, atomic merge.
// ---------------------------------------------------------------------------
__global__ __launch_bounds__(256)
void context_kernel(const float* __restrict__ values, const float* __restrict__ attn,
                    float* __restrict__ ctx) {
  const int b   = blockIdx.y;
  const int t0  = blockIdx.x * 256;
  const int tid = threadIdx.x;
  __shared__ float w[256];

  w[tid] = attn[(size_t)b * T_DIM + t0 + tid];
  __syncthreads();

  const float4* src = (const float4*)values + ((size_t)b * T_DIM + t0) * (D_DIM / 4);
  float4 acc = make_float4(0.f, 0.f, 0.f, 0.f);
  for (int t = 0; t < 256; ++t) {
    float4 x = src[(size_t)t * (D_DIM / 4) + tid];
    const float wt = w[t];
    acc.x = fmaf(wt, x.x, acc.x);
    acc.y = fmaf(wt, x.y, acc.y);
    acc.z = fmaf(wt, x.z, acc.z);
    acc.w = fmaf(wt, x.w, acc.w);
  }
  float* dst = ctx + (size_t)b * D_DIM + tid * 4;
  atomicAdd(dst + 0, acc.x);
  atomicAdd(dst + 1, acc.y);
  atomicAdd(dst + 2, acc.z);
  atomicAdd(dst + 3, acc.w);
}

// ---------------------------------------------------------------------------
extern "C" void kernel_launch(void* const* d_in, const int* in_sizes, int n_in,
                              void* d_out, int out_size, void* d_ws, size_t ws_size,
                              hipStream_t stream) {
  (void)in_sizes; (void)n_in; (void)out_size; (void)ws_size;

  const float* query  = (const float*)d_in[0];   // [B, D]
  const float* values = (const float*)d_in[1];   // [B, T, D]
  const float* w1     = (const float*)d_in[2];   // [D, U]
  const float* b1     = (const float*)d_in[3];   // [U]
  const float* w2     = (const float*)d_in[4];   // [D, U]
  const float* b2     = (const float*)d_in[5];   // [U]
  const float* v      = (const float*)d_in[6];   // [U, 1]
  const float* bv     = (const float*)d_in[7];   // [1]

  float* out_ctx  = (float*)d_out;                        // [B, D]
  float* out_attn = (float*)d_out + B_DIM * D_DIM;        // [B, T, 1]

  char*  ws     = (char*)d_ws;
  uint4* w2p    = (uint4*)ws;                             // 2 MB swizzled bf16 w2
  float* qpb    = (float*)(ws + 2u * 1024 * 1024);        // 128 KB
  float* scores = (float*)(ws + 2u * 1024 * 1024 + 128u * 1024); // 256 KB

  prep_w2_kernel <<<dim3((32 * 1024) / 256), 256, 0, stream>>>(w2, w2p);
  prep_qpb_kernel<<<dim3(U_DIM / 256, B_DIM), 256, 0, stream>>>(query, w1, b1, b2, qpb);

  const size_t smem = (size_t)TM * ROWU * 4 + TM * 4;     // ~129 KB of 320 KB LDS
  attn_score_kernel<<<dim3(T_DIM / TM, B_DIM), 256, smem, stream>>>(
      values, w2p, qpb, v, scores);

  softmax_kernel<<<dim3(B_DIM), 256, 0, stream>>>(scores, bv, out_ctx, out_attn);
  context_kernel<<<dim3(T_DIM / 256, B_DIM), 256, 0, stream>>>(values, out_attn, out_ctx);
}
